// HyKTNet_37391985279185
// MI455X (gfx1250) — compile-verified, matrix operand order checked by reading, not run on hardware
//
#include <hip/hip_runtime.h>

typedef __attribute__((ext_vector_type(16))) _Float16 v16h;
typedef __attribute__((ext_vector_type(8)))  float    v8f;

#define NV   11965   // N_E + 1
#define NC   188
#define NC1  189
#define ND1  101
#define DH   128
#define BB   128
#define SS_  500
#define MM   24000
#define BS   (BB*SS_)

#ifndef __has_builtin
#define __has_builtin(x) 0
#endif
#if defined(__AMDGCN__) && __has_builtin(__builtin_amdgcn_tensor_load_to_lds)
#define USE_TDM 1
#else
#define USE_TDM 0
#endif

union Frag { v16h v; unsigned u[8]; };

// K-index mapping for 16-bit A/B fragments (16x32 per wave32), per CDNA5 ISA 7.12.2:
// VGPR j (j<4): K = 2j + 8*half ; VGPR j (j>=4): K = 16 + 2(j-4) + 8*half
__device__ __forceinline__ int kmap(int j, int h) {
  return (j < 4) ? (2*j + 8*h) : (16 + 2*(j-4) + 8*h);
}
__device__ __forceinline__ float sigm(float x) { return 1.f / (1.f + __expf(-x)); }

// Build an A fragment (16x32 f16) from a 16x128 LDS tile
__device__ __forceinline__ Frag ldsA(const _Float16* base, int l15, int half, int kf) {
  Frag a;
#pragma unroll
  for (int j = 0; j < 8; ++j) {
    int kk = kf * 32 + kmap(j, half);
    a.u[j] = *(const unsigned*)(base + l15 * DH + kk);
  }
  return a;
}

#if USE_TDM
typedef unsigned int tdm_u32x4 __attribute__((ext_vector_type(4)));
typedef int          tdm_i32x8 __attribute__((ext_vector_type(8)));
typedef int          tdm_i32x4 __attribute__((ext_vector_type(4)));

// DMA a 2D f16 tile (tile_dim1=128 rows x tile_dim0=32 elems) global->LDS.
// rows_avail < 128 => TDM zero-fills OOB rows (reads beyond tensor dim return 0).
__device__ __forceinline__ void tdm_load_2d_f16(
    const _Float16* gsrc, unsigned lds_byte_off,
    unsigned rows_avail, unsigned row_stride_elems)
{
  unsigned long long ga = (unsigned long long)(uintptr_t)gsrc;
  tdm_u32x4 g0;
  g0[0] = 1u;                                                   // count=1, user D#
  g0[1] = lds_byte_off;                                         // lds_addr [63:32]
  g0[2] = (unsigned)ga;                                         // global_addr lo
  g0[3] = (unsigned)((ga >> 32) & 0x01FFFFFFull) | (2u << 30);  // addr hi | type=2
  tdm_i32x8 g1;
  g1[0] = 1 << 16;                                   // wg_mask=0, data_size=2B
  g1[1] = (int)(32u << 16);                          // abar=0 | tensor_dim0=32 lo16
  g1[2] = (int)((rows_avail & 0xFFFFu) << 16);       // dim0 hi=0 | tensor_dim1 lo16
  g1[3] = (int)(((rows_avail >> 16) & 0xFFFFu) | (32u << 16)); // dim1 hi | tile_dim0=32
  g1[4] = 128;                                       // tile_dim1=128 | tile_dim2=0
  g1[5] = (int)row_stride_elems;                     // tensor_dim0_stride lo32
  g1[6] = 0;                                         // stride hi | dim1_stride lo
  g1[7] = 0;
  tdm_i32x4 z4 = {0, 0, 0, 0};
#if __clang_major__ >= 23
  tdm_i32x8 z8 = {0, 0, 0, 0, 0, 0, 0, 0};
  __builtin_amdgcn_tensor_load_to_lds(g0, g1, z4, z4, z8, 0);
#else
  __builtin_amdgcn_tensor_load_to_lds(g0, g1, z4, z4, 0);
#endif
}
#endif

// cooperative fallback staging of one B tile (128 x 32 f16) into LDS
__device__ __forceinline__ void stage_copy(_Float16* dst, const _Float16* W, int ldw,
                                           int nblk, int N, int k0, int tid) {
  int n = tid >> 1;
  int ks = (tid & 1) * 16;
  int gn = nblk + n;
  unsigned* d = (unsigned*)(dst + n * 32 + ks);
  if (gn < N) {
    const unsigned* s = (const unsigned*)(W + (size_t)gn * ldw + k0 + ks);
#pragma unroll
    for (int q = 0; q < 8; ++q) d[q] = s[q];
  } else {
#pragma unroll
    for (int q = 0; q < 8; ++q) d[q] = 0u;
  }
}

// ---------------- generic WMMA GEMM: C = epi( A(MxK,f16) @ W(NxK,f16)^T ) ----------
// epi: y = acc*alpha + bias[n]; if g: y = y*g[n]+be[n]; if relu: max(0,y)
// B tiles double-buffered through LDS via TDM (tensor_load_to_lds) when available.
__global__ __launch_bounds__(256) void gemm_f16_wmma(
    const _Float16* __restrict__ A, int lda,
    const _Float16* __restrict__ W, int ldw,
    float* __restrict__ C, int ldc,
    int M, int N, int K, float alpha,
    const float* __restrict__ bias, const float* __restrict__ gam,
    const float* __restrict__ bet, int relu)
{
  __shared__ __align__(16) _Float16 Bs[2][128 * 32];
  int tid = threadIdx.x, wave = tid >> 5, lane = tid & 31;
  int l15 = lane & 15, half = lane >> 4;
  int m0 = blockIdx.x * 128 + wave * 16;
  int nblk = blockIdx.y * 128;
  int nb = K >> 5;                       // K is always a multiple of 32 here
  unsigned rows_avail = (unsigned)(N - nblk);

  v8f acc[8];
#pragma unroll
  for (int t = 0; t < 8; ++t)
#pragma unroll
    for (int r = 0; r < 8; ++r) acc[t][r] = 0.f;

#if USE_TDM
  bool issuer = (__builtin_amdgcn_readfirstlane((int)threadIdx.x) < 32);
  if (issuer)
    tdm_load_2d_f16(W + (size_t)nblk * ldw,
                    (unsigned)(uintptr_t)&Bs[0][0], rows_avail, (unsigned)ldw);
#else
  stage_copy(&Bs[0][0], W, ldw, nblk, N, 0, tid);
#endif

  for (int it = 0; it < nb; ++it) {
    int k0 = it << 5;
    const _Float16* Bcur = &Bs[it & 1][0];
#if USE_TDM
    if (issuer) __builtin_amdgcn_s_wait_tensorcnt(0);
    __syncthreads();
    if (issuer && (it + 1) < nb)       // overlap next tile DMA with this tile's WMMAs
      tdm_load_2d_f16(W + (size_t)nblk * ldw + (k0 + 32),
                      (unsigned)(uintptr_t)&Bs[(it + 1) & 1][0],
                      rows_avail, (unsigned)ldw);
#else
    __syncthreads();
#endif

    Frag a;
    int m = m0 + l15;
#pragma unroll
    for (int j = 0; j < 8; ++j) {
      int kk = k0 + kmap(j, half);
      a.u[j] = (m < M) ? *(const unsigned*)(A + (size_t)m * lda + kk) : 0u;
    }
    // preload all 8 B fragments, then issue WMMAs back-to-back
    Frag bfr[8];
#pragma unroll
    for (int t = 0; t < 8; ++t) {
      int nn = t * 16 + l15;
#pragma unroll
      for (int j = 0; j < 8; ++j)
        bfr[t].u[j] = *(const unsigned*)(Bcur + nn * 32 + kmap(j, half));
    }
#pragma unroll
    for (int t = 0; t < 8; ++t)
      acc[t] = __builtin_amdgcn_wmma_f32_16x16x32_f16(
          false, a.v, false, bfr[t].v, (short)0, acc[t], false, false);

#if !USE_TDM
    if (it + 1 < nb) stage_copy(&Bs[(it + 1) & 1][0], W, ldw, nblk, N, k0 + 32, tid);
#endif
  }

  for (int t = 0; t < 8; ++t) {
    int n = nblk + t * 16 + l15;
    if (n >= N) continue;
    float bn = bias ? bias[n] : 0.f;
    float gn = gam ? gam[n] : 1.f;
    float en = gam ? bet[n] : 0.f;
#pragma unroll
    for (int r = 0; r < 8; ++r) {
      int m = m0 + r + 8 * half;
      if (m >= M) continue;
      float y = acc[t][r] * alpha + bn;
      y = y * gn + en;
      if (relu) y = fmaxf(y, 0.f);
      C[(size_t)m * ldc + n] = y;
    }
  }
}

// ---------------- small helper kernels -------------------------------------------
__global__ void fzero_k(float* p, size_t n) {
  size_t i = (size_t)blockIdx.x * blockDim.x + threadIdx.x;
  if (i < n) p[i] = 0.f;
}
__global__ void cvt_h_k(const float* s, _Float16* d, size_t n) {
  size_t i = (size_t)blockIdx.x * blockDim.x + threadIdx.x;
  if (i < n) d[i] = (_Float16)s[i];
}
// dst[d][r] = (r<R) ? src[r][d] : 0   (dst is 128 x LD, f16)
__global__ void tpose_pad_k(const float* src, _Float16* dst, int R, int LD) {
  int i = blockIdx.x * blockDim.x + threadIdx.x;
  if (i >= 128 * LD) return;
  int d = i / LD, r = i % LD;
  dst[i] = (r < R) ? (_Float16)src[(size_t)r * DH + d] : (_Float16)0.f;
}
__global__ void count_k(const int* he, const int* hv, float* ecnt, float* vcnt) {
  int i = blockIdx.x * blockDim.x + threadIdx.x;
  if (i >= MM) return;
  atomicAdd(&ecnt[he[i]], 1.f);
  atomicAdd(&vcnt[hv[i]], 1.f);
}
// O[dst[i]*128+d] += X[src[i]*128+d]
__global__ void seg_add_k(const int* dsti, const int* srci,
                          const float* X, float* O) {
  size_t t = (size_t)blockIdx.x * blockDim.x + threadIdx.x;
  if (t >= (size_t)MM * DH) return;
  int i = (int)(t >> 7), d = (int)(t & 127);
  atomicAdd(&O[(size_t)dsti[i] * DH + d], X[(size_t)srci[i] * DH + d]);
}
__global__ void div_cnt_k(float* X, const float* cnt, int rows, int relu, _Float16* Xh) {
  size_t t = (size_t)blockIdx.x * blockDim.x + threadIdx.x;
  if (t >= (size_t)rows * DH) return;
  int r = (int)(t >> 7);
  float v = X[t] / fmaxf(cnt[r], 1.f);
  if (relu) v = fmaxf(v, 0.f);
  X[t] = v;
  if (Xh) Xh[t] = (_Float16)v;
}
__global__ void add_k(const float* a, const float* b, float* o, _Float16* oh, size_t n) {
  size_t i = (size_t)blockIdx.x * blockDim.x + threadIdx.x;
  if (i >= n) return;
  float v = a[i] + b[i];
  o[i] = v;
  oh[i] = (_Float16)v;
}

// one wave per node row: softmax over 189 gathered scores, scatter into A_c/A_d
__global__ __launch_bounds__(256) void softmax_scatter_k(
    const float* __restrict__ full, const int* __restrict__ ec,
    const int* __restrict__ ecd, float* __restrict__ Ac, float* __restrict__ Ad)
{
  int wave = threadIdx.x >> 5, lane = threadIdx.x & 31;
  int e = blockIdx.x * 8 + wave;
  if (e >= NV) return;
  const float* frow = full + (size_t)e * 192;
  const int* erow = ec + (size_t)e * NC1;
  const int* drow = ecd + (size_t)e * NC1;
  float mx = -3.0e38f;
  for (int c = lane; c < NC1; c += 32) {
    int idx = erow[c];
    float s = (idx == 0) ? -1e23f : frow[idx];
    mx = fmaxf(mx, s);
  }
  for (int off = 16; off; off >>= 1) mx = fmaxf(mx, __shfl_xor(mx, off, 32));
  float sum = 0.f;
  for (int c = lane; c < NC1; c += 32) {
    int idx = erow[c];
    float s = (idx == 0) ? -1e23f : frow[idx];
    sum += __expf(s - mx);
  }
  for (int off = 16; off; off >>= 1) sum += __shfl_xor(sum, off, 32);
  float inv = 1.f / sum;
  for (int c = lane; c < NC1; c += 32) {
    int idx = erow[c];
    float s = (idx == 0) ? -1e23f : frow[idx];
    float p = __expf(s - mx) * inv;
    atomicAdd(&Ac[(size_t)e * 192 + idx], p);
    atomicAdd(&Ad[(size_t)e * DH + drow[c]], p);
  }
}

// build Xcat f16: [e2[e] | ed_w[ed] | ep_w[ep] | c_all[e] | cd_all[e]]  (64000 x 640)
__global__ void xcat_k(const int* e_d, const int* ed_d, const int* ep_d,
                       const float* e2, const float* edw, const float* epw,
                       const float* c_all, const float* cd_all, _Float16* X)
{
  size_t t = (size_t)blockIdx.x * blockDim.x + threadIdx.x;
  if (t >= (size_t)BS * 640) return;
  size_t row = t / 640;
  int col = (int)(t % 640);
  int seg = col >> 7, d = col & 127;
  float v;
  if (seg == 0)      v = e2[(size_t)e_d[row] * DH + d];
  else if (seg == 1) v = edw[(size_t)ed_d[row] * DH + d];
  else if (seg == 2) v = epw[(size_t)ep_d[row] * DH + d];
  else if (seg == 3) v = c_all[(size_t)e_d[row] * DH + d];
  else               v = cd_all[(size_t)e_d[row] * DH + d];
  X[t] = (_Float16)v;
}

// out[r][d] = sum_k emb[r][k] * W[d][koff+k] (+bias[d]);   launch <<<R, DOUT>>>
__global__ void table_k(const float* emb, const float* W, int ldw, int koff,
                        const float* bias, float* out, int DOUT)
{
  int r = blockIdx.x, d = threadIdx.x;
  float s = bias ? bias[d] : 0.f;
  for (int k = 0; k < DH; ++k)
    s += emb[(size_t)r * DH + k] * W[(size_t)d * ldw + koff + k];
  out[(size_t)r * DOUT + d] = s;
}

// P1 += Tpp_ha[ha]+Tpp_ca[ca]; L1 += Tl_as[as]+Tl_ha[ha]+Tl_ca[ca]; G1 += ...
__global__ void addlut_k(float* P1, float* L1, float* G1,
                         const int* ha, const int* ca, const int* as_,
                         const float* Tpph, const float* Tppc,
                         const float* Tla, const float* Tlh, const float* Tlc,
                         const float* Tga, const float* Tgh, const float* Tgc)
{
  size_t t = (size_t)blockIdx.x * blockDim.x + threadIdx.x;
  if (t >= (size_t)BS * DH) return;
  size_t row = t >> 7;
  int d = (int)(t & 127);
  int h = ha[row], c = ca[row], a = as_[row];
  P1[t] += Tpph[h * DH + d] + Tppc[c * DH + d];
  L1[t] += Tla[a * DH + d] + Tlh[h * DH + d] + Tlc[c * DH + d];
  G1[t] += Tga[a * DH + d] + Tgh[h * DH + d] + Tgc[c * DH + d];
}

// ps output: s_pred depends only on small tables
__global__ void spred_k(const int* ha, const int* ca, const int* as_, const int* it,
                        const float* Sha, const float* Sca, const float* Sas,
                        const float* Sit, const float* bps, float* outps)
{
  int t = blockIdx.x * blockDim.x + threadIdx.x;
  if (t >= BS) return;
  float v = Sha[ha[t]] + Sca[ca[t]] + Sas[as_[t]] + Sit[it[t]] + bps[0];
  outps[t] = sigm(v);
}

// ---------------- persistent scan: 8 blocks x 16 batch rows, 500 steps -----------
__global__ __launch_bounds__(256) void scan_k(
    const float* __restrict__ h0,
    const _Float16* __restrict__ Wfh,   // 128x256, h-block at col 0
    const _Float16* __restrict__ Wpph,  // 128x512, h-block at col 128
    const _Float16* __restrict__ Wlh,   // 128x768, h-block at col 128
    const _Float16* __restrict__ Wlgh,  // 128x768, h-block at col 128
    const float* __restrict__ Tf_it, const float* __restrict__ Tl_a,
    const float* __restrict__ Tlg_a,
    const float* __restrict__ Wp, const float* __restrict__ bp,
    const float* __restrict__ P1, const float* __restrict__ L1,
    const float* __restrict__ G1,
    const int* __restrict__ it_data, const int* __restrict__ a_data,
    const float* __restrict__ ps, float* __restrict__ pm)
{
  __shared__ float h32[16 * DH], hf32[16 * DH], pre32[16 * DH];
  __shared__ _Float16 hh[16 * DH], hfh[16 * DH];
  __shared__ float WpL[DH], TfL[8 * DH], TlaL[2 * DH], TlgaL[2 * DH];
  __shared__ float alphaL[16];
  __shared__ int itL[16], aL[16];

  int tid = threadIdx.x, wave = tid >> 5, lane = tid & 31;
  int l15 = lane & 15, half = lane >> 4;
  int rb = blockIdx.x * 16;
  int nn = wave * 16 + l15;

  // register-resident B fragments for the 4 h-weight blocks (constant over steps)
  Frag bWf[4], bWpp[4], bWl[4], bWlg[4];
#pragma unroll
  for (int kf = 0; kf < 4; ++kf)
#pragma unroll
    for (int j = 0; j < 8; ++j) {
      int kk = kf * 32 + kmap(j, half);
      bWf[kf].u[j]  = *(const unsigned*)(Wfh  + (size_t)nn * 256 + 0   + kk);
      bWpp[kf].u[j] = *(const unsigned*)(Wpph + (size_t)nn * 512 + 128 + kk);
      bWl[kf].u[j]  = *(const unsigned*)(Wlh  + (size_t)nn * 768 + 128 + kk);
      bWlg[kf].u[j] = *(const unsigned*)(Wlgh + (size_t)nn * 768 + 128 + kk);
    }

  for (int t = tid; t < 16 * DH; t += 256) {
    float v = h0[(size_t)(rb + t / DH) * DH + (t & 127)];
    h32[t] = v;
    hh[t] = (_Float16)v;
  }
  for (int t = tid; t < 8 * DH; t += 256) TfL[t] = Tf_it[t];
  for (int t = tid; t < 2 * DH; t += 256) { TlaL[t] = Tl_a[t]; TlgaL[t] = Tlg_a[t]; }
  if (tid < DH) WpL[tid] = Wp[tid];
  float bpv = bp[0];
  __syncthreads();

  for (int st = 0; st < SS_; ++st) {
    if (tid < 16) {
      itL[tid] = it_data[(size_t)(rb + tid) * SS_ + st];
      aL[tid]  = a_data[(size_t)(rb + tid) * SS_ + st];
      if (st + 1 < SS_) {  // global_prefetch_b8 next step's streamed rows
        size_t nb = ((size_t)(rb + tid) * SS_ + (st + 1)) * DH;
        __builtin_prefetch(P1 + nb, 0, 0);
        __builtin_prefetch(L1 + nb, 0, 0);
        __builtin_prefetch(G1 + nb, 0, 0);
      }
    }
    __syncthreads();

    // fg = sigmoid(h @ Wf_h^T + Tf_it[it])  (1 where it<=1);  h_f = fg * h
    {
      Frag ah[4];
#pragma unroll
      for (int kf = 0; kf < 4; ++kf) ah[kf] = ldsA(hh, l15, half, kf);
      v8f accf = {};
#pragma unroll
      for (int kf = 0; kf < 4; ++kf)
        accf = __builtin_amdgcn_wmma_f32_16x16x32_f16(
            false, ah[kf].v, false, bWf[kf].v, (short)0, accf, false, false);
#pragma unroll
      for (int r = 0; r < 8; ++r) {
        int m = r + 8 * half;
        int itv = itL[m];
        float fg = (itv > 1) ? sigm(accf[r] + TfL[itv * DH + nn]) : 1.f;
        float hf = fg * h32[m * DH + nn];
        hf32[m * DH + nn] = hf;
        hfh[m * DH + nn] = (_Float16)hf;
      }
    }
    __syncthreads();

    // pre = relu(P1 + h_f @ Wpp_h^T)
    Frag af[4];
#pragma unroll
    for (int kf = 0; kf < 4; ++kf) af[kf] = ldsA(hfh, l15, half, kf);
    {
      v8f accp = {};
#pragma unroll
      for (int kf = 0; kf < 4; ++kf)
        accp = __builtin_amdgcn_wmma_f32_16x16x32_f16(
            false, af[kf].v, false, bWpp[kf].v, (short)0, accp, false, false);
#pragma unroll
      for (int r = 0; r < 8; ++r) {
        int m = r + 8 * half;
        size_t go = ((size_t)(rb + m) * SS_ + st) * DH + nn;
        pre32[m * DH + nn] = fmaxf(accp[r] + P1[go], 0.f);
      }
    }
    __syncthreads();

    // m_pred per row (16 threads/row, shuffle-reduce); alpha = a?max(m-s,0.4):1
    {
      int row = tid >> 4, sub = tid & 15;
      const float* pr = pre32 + row * DH;
      float s = 0.f;
#pragma unroll
      for (int k = 0; k < 8; ++k) s += pr[sub + 16 * k] * WpL[sub + 16 * k];
      s += __shfl_xor(s, 1, 32);
      s += __shfl_xor(s, 2, 32);
      s += __shfl_xor(s, 4, 32);
      s += __shfl_xor(s, 8, 32);
      if (sub == 0) {
        float m_ = sigm(s + bpv);
        pm[(size_t)(rb + row) * SS_ + st] = m_;
        float sp = ps[(size_t)(rb + row) * SS_ + st];
        alphaL[row] = aL[row] ? fmaxf(m_ - sp, 0.4f) : 1.f;
      }
    }
    __syncthreads();

    // lg / gate / h update
    {
      v8f accl = {}, accg = {};
#pragma unroll
      for (int kf = 0; kf < 4; ++kf) {
        accl = __builtin_amdgcn_wmma_f32_16x16x32_f16(
            false, af[kf].v, false, bWl[kf].v, (short)0, accl, false, false);
        accg = __builtin_amdgcn_wmma_f32_16x16x32_f16(
            false, af[kf].v, false, bWlg[kf].v, (short)0, accg, false, false);
      }
#pragma unroll
      for (int r = 0; r < 8; ++r) {
        int m = r + 8 * half;
        size_t go = ((size_t)(rb + m) * SS_ + st) * DH + nn;
        int av = aL[m];
        float al = alphaL[m];
        float lg = sigm(accl[r] + L1[go] + al * TlaL[av * DH + nn]);
        float gt = sigm(accg[r] + G1[go] + al * TlgaL[av * DH + nn]);
        float hf = hf32[m * DH + nn];
        float hn = lg * (1.f - gt) + hf * gt;
        h32[m * DH + nn] = hn;
        hh[m * DH + nn] = (_Float16)hn;
      }
    }
    __syncthreads();
  }
}

// ---------------- host orchestration ---------------------------------------------
extern "C" void kernel_launch(void* const* d_in, const int* in_sizes, int n_in,
                              void* d_out, int out_size, void* d_ws, size_t ws_size,
                              hipStream_t stream) {
  const int* e_data  = (const int*)d_in[0];
  const int* ed_data = (const int*)d_in[1];
  const int* ep_data = (const int*)d_in[2];
  const int* a_data  = (const int*)d_in[3];
  const int* as_data = (const int*)d_in[4];
  const int* ha_data = (const int*)d_in[5];
  const int* ca_data = (const int*)d_in[6];
  const int* it_data = (const int*)d_in[7];
  const int* ec_m    = (const int*)d_in[8];
  const int* ecd_m   = (const int*)d_in[9];
  const int* hyper_v = (const int*)d_in[10];
  const int* hyper_e = (const int*)d_in[11];
  const float* deliver_h = (const float*)d_in[12];
  const float* e_w  = (const float*)d_in[13];
  const float* c_w  = (const float*)d_in[14];
  const float* ed_w = (const float*)d_in[15];
  const float* cd_w = (const float*)d_in[16];
  const float* a_w  = (const float*)d_in[17];
  const float* it_w = (const float*)d_in[18];
  const float* ep_w = (const float*)d_in[19];
  const float* ha_w = (const float*)d_in[20];
  const float* ca_w = (const float*)d_in[21];
  const float* as_w = (const float*)d_in[22];
  const float* c1W = (const float*)d_in[23]; const float* c1b = (const float*)d_in[24];
  const float* c1g = (const float*)d_in[25]; const float* c1e = (const float*)d_in[26];
  const float* c2W = (const float*)d_in[27]; const float* c2b = (const float*)d_in[28];
  const float* c2g = (const float*)d_in[29]; const float* c2e = (const float*)d_in[30];
  const float* Wi  = (const float*)d_in[31]; const float* bi  = (const float*)d_in[32];
  const float* Wps = (const float*)d_in[33]; const float* bps = (const float*)d_in[34];
  const float* Wf  = (const float*)d_in[35]; const float* bf  = (const float*)d_in[36];
  const float* Wpp = (const float*)d_in[37]; const float* bpp = (const float*)d_in[38];
  const float* Wp  = (const float*)d_in[39]; const float* bp  = (const float*)d_in[40];
  const float* Wl  = (const float*)d_in[41]; const float* bl  = (const float*)d_in[42];
  const float* Wlg = (const float*)d_in[43]; const float* blg = (const float*)d_in[44];

  float* outps = (float*)d_out;          // ps (B,S)
  float* outpm = (float*)d_out + BS;     // pm (B,S)

  char* wp = (char*)d_ws;
  auto alloc = [&](size_t bytes) -> void* {
    void* p = (void*)wp;
    wp += (bytes + 255) & ~(size_t)255;
    return p;
  };
  _Float16* ewh   = (_Float16*)alloc((size_t)NV * DH * 2);
  _Float16* W1h   = (_Float16*)alloc(DH * DH * 2);
  _Float16* W2h   = (_Float16*)alloc(DH * DH * 2);
  _Float16* cwh   = (_Float16*)alloc(NC1 * DH * 2);
  _Float16* cwTh  = (_Float16*)alloc(DH * 192 * 2);
  _Float16* cdwTh = (_Float16*)alloc(DH * DH * 2);
  _Float16* Wih   = (_Float16*)alloc(DH * 640 * 2);
  _Float16* Wfh   = (_Float16*)alloc(DH * 256 * 2);
  _Float16* Wpph  = (_Float16*)alloc(DH * 512 * 2);
  _Float16* Wlh   = (_Float16*)alloc(DH * 768 * 2);
  _Float16* Wlgh  = (_Float16*)alloc(DH * 768 * 2);
  float* ecnt = (float*)alloc(NC * 4);
  float* vcnt = (float*)alloc((size_t)NV * 4);
  float* Y    = (float*)alloc((size_t)NV * DH * 4);
  float* E    = (float*)alloc((size_t)NC * DH * 4);
  float* Xo   = (float*)alloc((size_t)NV * DH * 4);
  _Float16* Xoh = (_Float16*)alloc((size_t)NV * DH * 2);
  float* e2   = (float*)alloc((size_t)NV * DH * 4);
  _Float16* e2h = (_Float16*)alloc((size_t)NV * DH * 2);
  float* full = (float*)alloc((size_t)NV * 192 * 4);
  float* Ac   = (float*)alloc((size_t)NV * 192 * 4);
  float* Ad   = (float*)alloc((size_t)NV * DH * 4);
  _Float16* Ach = (_Float16*)alloc((size_t)NV * 192 * 2);
  _Float16* Adh = (_Float16*)alloc((size_t)NV * DH * 2);
  float* c_all  = (float*)alloc((size_t)NV * DH * 4);
  float* cd_all = (float*)alloc((size_t)NV * DH * 4);
  _Float16* Xcat = (_Float16*)alloc((size_t)BS * 640 * 2);
  float* iF32 = (float*)alloc((size_t)BS * DH * 4);
  _Float16* ih = (_Float16*)alloc((size_t)BS * DH * 2);
  float* Tf_it = (float*)alloc(8 * DH * 4);
  float* Tpph  = (float*)alloc(11 * DH * 4);
  float* Tppc  = (float*)alloc(11 * DH * 4);
  float* Tla   = (float*)alloc(2 * DH * 4);
  float* Tlas  = (float*)alloc(8 * DH * 4);
  float* Tlha  = (float*)alloc(11 * DH * 4);
  float* Tlca  = (float*)alloc(11 * DH * 4);
  float* Tga   = (float*)alloc(2 * DH * 4);
  float* Tgas  = (float*)alloc(8 * DH * 4);
  float* Tgha  = (float*)alloc(11 * DH * 4);
  float* Tgca  = (float*)alloc(11 * DH * 4);
  float* Sha = (float*)alloc(11 * 4);
  float* Sca = (float*)alloc(11 * 4);
  float* Sas = (float*)alloc(8 * 4);
  float* Sit = (float*)alloc(8 * 4);
  // reuse dead regions: Xcat (82MB) holds P1+L1 after i_all; iF32 holds G1 after cvt
  float* P1 = (float*)Xcat;
  float* L1 = (float*)((char*)Xcat + (size_t)BS * DH * 4);
  float* G1 = iF32;

  auto cvt = [&](const float* s, _Float16* d, size_t n) {
    cvt_h_k<<<dim3((unsigned)((n + 255) / 256)), 256, 0, stream>>>(s, d, n);
  };
  auto zero = [&](float* p, size_t n) {
    fzero_k<<<dim3((unsigned)((n + 255) / 256)), 256, 0, stream>>>(p, n);
  };

  // f16 conversions of inputs
  cvt(e_w, ewh, (size_t)NV * DH);
  cvt(c1W, W1h, DH * DH);
  cvt(c2W, W2h, DH * DH);
  cvt(c_w, cwh, NC1 * DH);
  cvt(Wi, Wih, DH * 640);
  cvt(Wf, Wfh, DH * 256);
  cvt(Wpp, Wpph, DH * 512);
  cvt(Wl, Wlh, DH * 768);
  cvt(Wlg, Wlgh, DH * 768);
  tpose_pad_k<<<dim3((128 * 192 + 255) / 256), 256, 0, stream>>>(c_w, cwTh, NC1, 192);
  tpose_pad_k<<<dim3((128 * 128 + 255) / 256), 256, 0, stream>>>(cd_w, cdwTh, ND1, 128);

  // incidence counts
  zero(ecnt, NC); zero(vcnt, NV);
  count_k<<<dim3((MM + 255) / 256), 256, 0, stream>>>(hyper_e, hyper_v, ecnt, vcnt);

  const unsigned gM = (NV + 127) / 128;
  const size_t segN = (size_t)MM * DH;
  const unsigned gSeg = (unsigned)((segN + 255) / 256);
  const unsigned gNVD = (unsigned)(((size_t)NV * DH + 255) / 256);

  // ---- conv layer 1 ----
  gemm_f16_wmma<<<dim3(gM, 1), 256, 0, stream>>>(ewh, DH, W1h, DH, Y, DH,
      NV, DH, DH, 1.f, c1b, c1g, c1e, 0);
  zero(E, (size_t)NC * DH);
  seg_add_k<<<dim3(gSeg), 256, 0, stream>>>(hyper_e, hyper_v, Y, E);
  div_cnt_k<<<dim3((NC * DH + 255) / 256), 256, 0, stream>>>(E, ecnt, NC, 0, nullptr);
  zero(Xo, (size_t)NV * DH);
  seg_add_k<<<dim3(gSeg), 256, 0, stream>>>(hyper_v, hyper_e, E, Xo);
  div_cnt_k<<<dim3(gNVD), 256, 0, stream>>>(Xo, vcnt, NV, 1, Xoh);

  // ---- conv layer 2 + residual ----
  gemm_f16_wmma<<<dim3(gM, 1), 256, 0, stream>>>(Xoh, DH, W2h, DH, Y, DH,
      NV, DH, DH, 1.f, c2b, c2g, c2e, 0);
  zero(E, (size_t)NC * DH);
  seg_add_k<<<dim3(gSeg), 256, 0, stream>>>(hyper_e, hyper_v, Y, E);
  div_cnt_k<<<dim3((NC * DH + 255) / 256), 256, 0, stream>>>(E, ecnt, NC, 0, nullptr);
  zero(Xo, (size_t)NV * DH);
  seg_add_k<<<dim3(gSeg), 256, 0, stream>>>(hyper_v, hyper_e, E, Xo);
  div_cnt_k<<<dim3(gNVD), 256, 0, stream>>>(Xo, vcnt, NV, 0, nullptr);
  add_k<<<dim3(gNVD), 256, 0, stream>>>(Xo, e_w, e2, e2h, (size_t)NV * DH);

  // ---- attention: scores -> softmax -> scatter -> weighted sums ----
  gemm_f16_wmma<<<dim3(gM, 2), 256, 0, stream>>>(e2h, DH, cwh, DH, full, 192,
      NV, NC1, DH, 1.f / 64.f, nullptr, nullptr, nullptr, 0);
  zero(Ac, (size_t)NV * 192); zero(Ad, (size_t)NV * DH);
  softmax_scatter_k<<<dim3((NV + 7) / 8), 256, 0, stream>>>(full, ec_m, ecd_m, Ac, Ad);
  cvt(Ac, Ach, (size_t)NV * 192);
  cvt(Ad, Adh, (size_t)NV * DH);
  gemm_f16_wmma<<<dim3(gM, 1), 256, 0, stream>>>(Ach, 192, cwTh, 192, c_all, DH,
      NV, DH, 192, 1.f, nullptr, nullptr, nullptr, 0);
  gemm_f16_wmma<<<dim3(gM, 1), 256, 0, stream>>>(Adh, DH, cdwTh, DH, cd_all, DH,
      NV, DH, DH, 1.f, nullptr, nullptr, nullptr, 0);

  // ---- input projection i_all and hoisted P1/L1/G1 streams ----
  xcat_k<<<dim3((unsigned)(((size_t)BS * 640 + 255) / 256)), 256, 0, stream>>>(
      e_data, ed_data, ep_data, e2, ed_w, ep_w, c_all, cd_all, Xcat);
  gemm_f16_wmma<<<dim3(BS / 128, 1), 256, 0, stream>>>(Xcat, 640, Wih, 640, iF32, DH,
      BS, DH, 640, 1.f, bi, nullptr, nullptr, 0);
  cvt(iF32, ih, (size_t)BS * DH);
  // Xcat/iF32 now dead -> reuse for P1/L1/G1
  gemm_f16_wmma<<<dim3(BS / 128, 1), 256, 0, stream>>>(ih, DH, Wpph, 512, P1, DH,
      BS, DH, DH, 1.f, bpp, nullptr, nullptr, 0);
  gemm_f16_wmma<<<dim3(BS / 128, 1), 256, 0, stream>>>(ih, DH, Wlh, 768, L1, DH,
      BS, DH, DH, 1.f, bl, nullptr, nullptr, 0);
  gemm_f16_wmma<<<dim3(BS / 128, 1), 256, 0, stream>>>(ih, DH, Wlgh, 768, G1, DH,
      BS, DH, DH, 1.f, blg, nullptr, nullptr, 0);

  // ---- small tables ----
  table_k<<<dim3(8), 128, 0, stream>>>(it_w, Wf, 256, 128, bf, Tf_it, DH);
  table_k<<<dim3(11), 128, 0, stream>>>(ha_w, Wpp, 512, 256, nullptr, Tpph, DH);
  table_k<<<dim3(11), 128, 0, stream>>>(ca_w, Wpp, 512, 384, nullptr, Tppc, DH);
  table_k<<<dim3(2), 128, 0, stream>>>(a_w, Wl, 768, 256, nullptr, Tla, DH);
  table_k<<<dim3(8), 128, 0, stream>>>(as_w, Wl, 768, 384, nullptr, Tlas, DH);
  table_k<<<dim3(11), 128, 0, stream>>>(ha_w, Wl, 768, 512, nullptr, Tlha, DH);
  table_k<<<dim3(11), 128, 0, stream>>>(ca_w, Wl, 768, 640, nullptr, Tlca, DH);
  table_k<<<dim3(2), 128, 0, stream>>>(a_w, Wlg, 768, 256, nullptr, Tga, DH);
  table_k<<<dim3(8), 128, 0, stream>>>(as_w, Wlg, 768, 384, nullptr, Tgas, DH);
  table_k<<<dim3(11), 128, 0, stream>>>(ha_w, Wlg, 768, 512, nullptr, Tgha, DH);
  table_k<<<dim3(11), 128, 0, stream>>>(ca_w, Wlg, 768, 640, nullptr, Tgca, DH);
  table_k<<<dim3(11), 1, 0, stream>>>(ha_w, Wps, 512, 0, nullptr, Sha, 1);
  table_k<<<dim3(11), 1, 0, stream>>>(ca_w, Wps, 512, 128, nullptr, Sca, 1);
  table_k<<<dim3(8), 1, 0, stream>>>(as_w, Wps, 512, 256, nullptr, Sas, 1);
  table_k<<<dim3(8), 1, 0, stream>>>(it_w, Wps, 512, 384, nullptr, Sit, 1);

  addlut_k<<<dim3((unsigned)(((size_t)BS * DH + 255) / 256)), 256, 0, stream>>>(
      P1, L1, G1, ha_data, ca_data, as_data,
      Tpph, Tppc, Tlas, Tlha, Tlca, Tgas, Tgha, Tgca);

  // ps output (s_pred) — h-independent
  spred_k<<<dim3((BS + 255) / 256), 256, 0, stream>>>(
      ha_data, ca_data, as_data, it_data, Sha, Sca, Sas, Sit, bps, outps);

  // ---- the sequential scan: 8 persistent blocks, register-resident WMMA weights ----
  scan_k<<<dim3(BB / 16), 256, 0, stream>>>(
      deliver_h, Wfh, Wpph, Wlh, Wlgh, Tf_it, Tla, Tga, Wp, bp,
      P1, L1, G1, it_data, a_data, outps, outpm);
}
